// FACL_45964740002204
// MI455X (gfx1250) — matrix operand order repaired and checked by reading, not compiled, not run on hardware
//
#include <hip/hip_runtime.h>
#include <hip/hip_bf16.h>
#include <math.h>

typedef __attribute__((ext_vector_type(16))) __bf16 v16bf;
typedef __attribute__((ext_vector_type(8)))  float  v8f;

struct U4 { unsigned int x, y, z, w; };
struct U8 { U4 lo, hi; };

#define BM 128
#define BN 128
#define BK 32
#define LDSR 40   // padded LDS row stride in bf16 elements (80B -> conflict-free)

// ---------------------------------------------------------------------------
// Tiled bf16 WMMA GEMM:  C[M][N] = A[M][Kp] * Bw[N][Kp]^T   (i.e. X @ W.T)
//   A  : bf16, row stride lda
//   Bw : bf16, weight in natural [N][Kp] row-major layout
// Staging uses GLOBAL_LOAD_ASYNC_TO_LDS_B128 (ASYNCcnt) with double-buffered
// LDS: tile k+1 streams in while WMMAs consume tile k; one barrier per step.
// Out-of-range A rows are clamped (their outputs are discarded in epilogue).
// Epilogue is specialized at compile time to avoid dead branches / spills:
//   RELU: clamp at 0; RES: += Res[row,col]; BIAS: += bias[col];
//   C2F: also write f32 copy C2; CBF: also write bf16 copy Cbf.
// ---------------------------------------------------------------------------
template<bool RELU, bool RES, bool BIAS, bool C2F, bool CBF>
__global__ __launch_bounds__(256)
void gemm_bf16_kernel(const __bf16* __restrict__ A, long lda,
                      const __bf16* __restrict__ Bw,
                      float* __restrict__ C, float* __restrict__ C2,
                      const float* __restrict__ Res,
                      const float* __restrict__ bias,
                      __bf16* __restrict__ Cbf,
                      int M, int Kp, long ldc)
{
  __shared__ __bf16 sA[2][BM * LDSR];
  __shared__ __bf16 sB[2][BN * LDSR];

  const int tid  = threadIdx.x;
  const int lane = tid & 31;
  const int wid  = tid >> 5;
  const int wm   = wid & 3;           // 4 wave-rows of 32
  const int wn   = wid >> 2;          // 2 wave-cols of 64
  const long bm  = (long)blockIdx.x * BM;
  const long bn  = (long)blockIdx.y * BN;

  const int la_row = tid >> 1;        // 0..127
  const int la_off = (tid & 1) * 16;  // 0 or 16 bf16 elements

  const int half = lane >> 4;
  const int l16  = lane & 15;

  // clamp OOB A rows to a valid row: their outputs are masked at the end
  long arow = bm + la_row; if (arow >= M) arow = M - 1;
  const __bf16* gA = A + arow * lda + la_off;
  const __bf16* gB = Bw + (bn + la_row) * (long)Kp + la_off;
  const unsigned ldsA0 = (unsigned)(uintptr_t)(&sA[0][la_row * LDSR + la_off]);
  const unsigned ldsA1 = (unsigned)(uintptr_t)(&sA[1][la_row * LDSR + la_off]);
  const unsigned ldsB0 = (unsigned)(uintptr_t)(&sB[0][la_row * LDSR + la_off]);
  const unsigned ldsB1 = (unsigned)(uintptr_t)(&sB[1][la_row * LDSR + la_off]);

  // async-stage one BK slice of A and B into LDS buffer `buf`
  auto stage = [&](int k0, int buf) {
    const __bf16* pa = gA + k0;
    const __bf16* pb = gB + k0;
    unsigned la = buf ? ldsA1 : ldsA0;
    unsigned lb = buf ? ldsB1 : ldsB0;
    asm volatile("global_load_async_to_lds_b128 %0, %1, off"
                 :: "v"(la), "v"(pa) : "memory");
    asm volatile("global_load_async_to_lds_b128 %0, %1, off offset:16"
                 :: "v"(la), "v"(pa) : "memory");
    asm volatile("global_load_async_to_lds_b128 %0, %1, off"
                 :: "v"(lb), "v"(pb) : "memory");
    asm volatile("global_load_async_to_lds_b128 %0, %1, off offset:16"
                 :: "v"(lb), "v"(pb) : "memory");
  };

  const v8f vzero = {0.f,0.f,0.f,0.f,0.f,0.f,0.f,0.f};
  v8f acc[2][4];
  for (int i = 0; i < 2; ++i)
    for (int j = 0; j < 4; ++j)
      acc[i][j] = vzero;

  // prologue: stage first tile
  stage(0, 0);
  asm volatile("s_wait_asynccnt 0" ::: "memory");
  __syncthreads();

  int cur = 0;
  for (int k0 = 0; k0 < Kp; k0 += BK) {
    if (k0 + BK < Kp) stage(k0 + BK, cur ^ 1);   // prefetch next tile (async)

    const __bf16* cA = sA[cur];
    const __bf16* cB = sB[cur];

    // A fragment: lane row = l16; elems 0..7 -> K=8*half+0..7, 8..15 -> K=16+8*half+0..7
    v16bf af[2];
    for (int sm = 0; sm < 2; ++sm) {
      const char* rp = (const char*)(cA + (wm * 32 + sm * 16 + l16) * LDSR);
      U8 t;
      t.lo = *(const U4*)(rp + 16 * half);
      t.hi = *(const U4*)(rp + 32 + 16 * half);
      af[sm] = __builtin_bit_cast(v16bf, t);
    }
    // B fragment: lane col = l16; elems 0..7 -> K=8*half+0..7, 8..15 -> K=16+8*half+0..7
    v16bf bfr[4];
    for (int sn = 0; sn < 4; ++sn) {
      const char* rp = (const char*)(cB + (wn * 64 + sn * 16 + l16) * LDSR);
      U8 t;
      t.lo = *(const U4*)(rp + 16 * half);
      t.hi = *(const U4*)(rp + 32 + 16 * half);
      bfr[sn] = __builtin_bit_cast(v16bf, t);
    }

    for (int sm = 0; sm < 2; ++sm)
      for (int sn = 0; sn < 4; ++sn)
        acc[sm][sn] = __builtin_amdgcn_wmma_f32_16x16x32_bf16(
            false, af[sm], false, bfr[sn], (short)0, acc[sm][sn], false, false);

    asm volatile("s_wait_asynccnt 0" ::: "memory");  // next tile fully in LDS
    __syncthreads();                                 // publish to all waves
    cur ^= 1;
  }

  // epilogue: D layout m = r + 8*half, n = l16
  for (int sm = 0; sm < 2; ++sm) {
    for (int sn = 0; sn < 4; ++sn) {
      const long gcol = bn + wn * 64 + sn * 16 + l16;
      float bv = 0.f;
      if (BIAS) bv = bias[gcol];
      for (int r = 0; r < 8; ++r) {
        const long grow = bm + wm * 32 + sm * 16 + r + 8 * half;
        if (grow < M) {
          const long o = grow * ldc + gcol;
          float v = acc[sm][sn][r];
          if (BIAS) v += bv;
          if (RES)  v += Res[o];
          if (RELU) v = v > 0.f ? v : 0.f;
          C[o] = v;
          if (C2F) C2[o]  = v;
          if (CBF) Cbf[o] = (__bf16)v;
        }
      }
    }
  }
}

// ---------------------------------------------------------------------------
// Elementwise / gather kernels
// ---------------------------------------------------------------------------
__global__ void convert_pad_kernel(const float* __restrict__ in, __bf16* __restrict__ out,
                                   long rows, int K, int Kp)
{
  long i = (long)blockIdx.x * blockDim.x + threadIdx.x;
  long total = rows * Kp;
  if (i >= total) return;
  long r = i / Kp;
  int  c = (int)(i - r * Kp);
  out[i] = (c < K) ? (__bf16)in[r * K + c] : (__bf16)0.f;
}

// message_atom[i] += sum_j(mb[a2b]) * max_j(mb[a2b])
__global__ void agg_accum_kernel(const float* __restrict__ mb, const int* __restrict__ a2b,
                                 float* __restrict__ matom, long total)
{
  long i = (long)blockIdx.x * blockDim.x + threadIdx.x;
  if (i >= total) return;
  long row = i >> 9;
  int  col = (int)(i & 511);
  const int* idx = a2b + row * 6;
  float s = 0.f, mx = -3.402823466e38f;
#pragma unroll
  for (int j = 0; j < 6; ++j) {
    float v = mb[(long)idx[j] * 512 + col];
    s += v;
    mx = v > mx ? v : mx;
  }
  matom[i] += s * mx;
}

// cat = bf16([agg, message_atom, input_atom])  (row stride 1536)
__global__ void final_cat_kernel(const float* __restrict__ mb, const int* __restrict__ a2b,
                                 const float* __restrict__ matom, const float* __restrict__ iatom,
                                 __bf16* __restrict__ cat, long total)
{
  long i = (long)blockIdx.x * blockDim.x + threadIdx.x;
  if (i >= total) return;
  long row = i >> 9;
  int  col = (int)(i & 511);
  const int* idx = a2b + row * 6;
  float s = 0.f, mx = -3.402823466e38f;
#pragma unroll
  for (int j = 0; j < 6; ++j) {
    float v = mb[(long)idx[j] * 512 + col];
    s += v;
    mx = v > mx ? v : mx;
  }
  __bf16* c = cat + row * 1536;
  c[col]        = (__bf16)(s * mx);
  c[512 + col]  = (__bf16)matom[i];
  c[1024 + col] = (__bf16)iatom[i];
}

// pre_bf = bf16(message_atom[b2a] - message_bond[b2revb])
__global__ void bond_pre_kernel(const float* __restrict__ matom, const float* __restrict__ mb,
                                const int* __restrict__ b2a, const int* __restrict__ b2revb,
                                __bf16* __restrict__ pre, long total)
{
  long i = (long)blockIdx.x * blockDim.x + threadIdx.x;
  if (i >= total) return;
  long b = i >> 9;
  int  col = (int)(i & 511);
  float v = matom[(long)b2a[b] * 512 + col] - mb[(long)b2revb[b] * 512 + col];
  pre[i] = (__bf16)v;
}

// message = relu(node + gru_bias); row0 -> pad row of gruc (both halves),
// rows 1.. -> bf16 GRU input sequence
__global__ void postnode_kernel(const float* __restrict__ node, const float* __restrict__ gbias,
                                __bf16* __restrict__ msg_bf, __bf16* __restrict__ gruc, long total)
{
  long i = (long)blockIdx.x * blockDim.x + threadIdx.x;
  if (i >= total) return;
  long row = i >> 9;
  int  col = (int)(i & 511);
  float m = node[i] + gbias[col];
  m = m > 0.f ? m : 0.f;
  if (row == 0) {
    gruc[col]       = (__bf16)m;
    gruc[512 + col] = (__bf16)m;
  } else {
    msg_bf[i - 512] = (__bf16)m;
  }
}

// h0 = max over 48 atoms of node (f32 + bf16, for both directions)
__global__ void h0_kernel(const float* __restrict__ node, float* __restrict__ hf,
                          float* __restrict__ hb, __bf16* __restrict__ hbf_f,
                          __bf16* __restrict__ hbf_b, long total)
{
  long i = (long)blockIdx.x * blockDim.x + threadIdx.x;
  if (i >= total) return;
  long mol = i >> 9;
  int  col = (int)(i & 511);
  const float* p = node + (1 + mol * 48) * 512 + col;
  float mx = -3.402823466e38f;
#pragma unroll 4
  for (int a = 0; a < 48; ++a) mx = fmaxf(mx, p[(long)a * 512]);
  hf[i] = mx; hb[i] = mx;
  hbf_f[i] = (__bf16)mx; hbf_b[i] = (__bf16)mx;
}

// torch-style GRU cell combine; writes new h (f32 + bf16) and output slice
__global__ void gru_combine_kernel(const float* __restrict__ gi, const float* __restrict__ gh,
                                   float* __restrict__ h, __bf16* __restrict__ hbf,
                                   __bf16* __restrict__ gruc, int t, int doff, long total)
{
  long i = (long)blockIdx.x * blockDim.x + threadIdx.x;
  if (i >= total) return;
  long mol = i >> 9;
  int  j   = (int)(i & 511);
  long g = mol * 1536;
  float ir = gi[g + j], iz = gi[g + 512 + j], in = gi[g + 1024 + j];
  float hr = gh[g + j], hz = gh[g + 512 + j], hn = gh[g + 1024 + j];
  float r = 1.f / (1.f + __expf(-(ir + hr)));
  float z = 1.f / (1.f + __expf(-(iz + hz)));
  float n = tanhf(in + r * hn);
  float hv = (1.f - z) * n + z * h[i];
  h[i]   = hv;
  hbf[i] = (__bf16)hv;
  gruc[(long)(1 + mol * 48 + t) * 1024 + doff + j] = (__bf16)hv;
}

// ---------------------------------------------------------------------------
// Host orchestration
// ---------------------------------------------------------------------------
template<bool RELU, bool RES, bool BIAS, bool C2F, bool CBF>
static inline void launch_gemm_t(hipStream_t s, const __bf16* A, long lda, const __bf16* Bw,
                                 float* C, float* C2, const float* Res, const float* bias,
                                 __bf16* Cbf, int M, int N, int Kp, long ldc)
{
  dim3 grid((M + BM - 1) / BM, N / BN);
  gemm_bf16_kernel<RELU, RES, BIAS, C2F, CBF>
      <<<grid, dim3(256), 0, s>>>(A, lda, Bw, C, C2, Res, bias, Cbf, M, Kp, ldc);
}

static inline void launch_conv(hipStream_t s, const float* in, __bf16* out,
                               long rows, int K, int Kp)
{
  long total = rows * Kp;
  convert_pad_kernel<<<(unsigned)((total + 255) / 256), 256, 0, s>>>(in, out, rows, K, Kp);
}

extern "C" void kernel_launch(void* const* d_in, const int* in_sizes, int n_in,
                              void* d_out, int out_size, void* d_ws, size_t ws_size,
                              hipStream_t stream)
{
  const float* f_atoms  = (const float*)d_in[0];
  const float* f_bonds  = (const float*)d_in[1];
  const float* W_i_atom = (const float*)d_in[2];
  const float* W_i_bond = (const float*)d_in[3];
  const float* W_h      = (const float*)d_in[4];
  const float* W_lr     = (const float*)d_in[5];
  const float* gru_bias = (const float*)d_in[6];
  const float* Wih_f    = (const float*)d_in[7];
  const float* Whh_f    = (const float*)d_in[8];
  const float* bih_f    = (const float*)d_in[9];
  const float* bhh_f    = (const float*)d_in[10];
  const float* Wih_b    = (const float*)d_in[11];
  const float* Whh_b    = (const float*)d_in[12];
  const float* bih_b    = (const float*)d_in[13];
  const float* bhh_b    = (const float*)d_in[14];
  const float* W_o      = (const float*)d_in[15];
  const float* b_o      = (const float*)d_in[16];
  const int*   a2b      = (const int*)d_in[17];
  const int*   b2a      = (const int*)d_in[18];
  const int*   b2revb   = (const int*)d_in[19];

  const int NA = in_sizes[0] / 133;       // 24577
  const int NB = in_sizes[1] / 147;       // 51201
  const int n_mols = (NA - 1) / 48;       // 512
  const long NAH = (long)NA * 512;
  const long NBH = (long)NB * 512;
  const long MH  = (long)n_mols * 512;

  // ---- workspace layout (bump allocator, 256B aligned) ----
  char* p = (char*)d_ws;
  auto alloc = [&](size_t bytes) -> char* {
    char* r = p; p += (bytes + 255) & ~(size_t)255; return r;
  };
  __bf16* fa_bf   = (__bf16*)alloc((size_t)NA * 160 * 2);
  __bf16* fb_bf   = (__bf16*)alloc((size_t)NB * 160 * 2);
  __bf16* WbA     = (__bf16*)alloc((size_t)512 * 160 * 2);
  __bf16* WbBnd   = (__bf16*)alloc((size_t)512 * 160 * 2);
  __bf16* WbH     = (__bf16*)alloc((size_t)5 * 512 * 512 * 2);
  __bf16* WbLR    = (__bf16*)alloc((size_t)512 * 1536 * 2);
  __bf16* WbIHf   = (__bf16*)alloc((size_t)1536 * 512 * 2);
  __bf16* WbHHf   = (__bf16*)alloc((size_t)1536 * 512 * 2);
  __bf16* WbIHb   = (__bf16*)alloc((size_t)1536 * 512 * 2);
  __bf16* WbHHb   = (__bf16*)alloc((size_t)1536 * 512 * 2);
  __bf16* WbO     = (__bf16*)alloc((size_t)512 * 1024 * 2);
  float*  iatom   = (float*)alloc((size_t)NAH * 4);
  float*  ibond   = (float*)alloc((size_t)NBH * 4);
  float*  matom   = (float*)alloc((size_t)NAH * 4);
  float*  mbond   = (float*)alloc((size_t)NBH * 4);
  __bf16* pre_bf  = (__bf16*)alloc((size_t)NBH * 2);
  __bf16* cat_bf  = (__bf16*)alloc((size_t)NA * 1536 * 2);
  float*  node    = (float*)alloc((size_t)NAH * 4);
  __bf16* msg_bf  = (__bf16*)alloc((size_t)(NA - 1) * 512 * 2);
  __bf16* gruc_bf = (__bf16*)alloc((size_t)NA * 1024 * 2);
  float*  h_f     = (float*)alloc((size_t)MH * 4);
  float*  h_b     = (float*)alloc((size_t)MH * 4);
  __bf16* hbf_f   = (__bf16*)alloc((size_t)MH * 2);
  __bf16* hbf_b   = (__bf16*)alloc((size_t)MH * 2);
  float*  gi      = (float*)alloc((size_t)n_mols * 1536 * 4);
  float*  gh      = (float*)alloc((size_t)n_mols * 1536 * 4);
  (void)ws_size; (void)n_in; (void)out_size;

  const unsigned TB = 256;
  const unsigned gNA = (unsigned)((NAH + TB - 1) / TB);
  const unsigned gNB = (unsigned)((NBH + TB - 1) / TB);
  const unsigned gM  = (unsigned)((MH + TB - 1) / TB);

  // ---- bf16 conversions (activations padded K->160; weights natural layout) ----
  launch_conv(stream, f_atoms,  fa_bf, NA, 133, 160);
  launch_conv(stream, f_bonds,  fb_bf, NB, 147, 160);
  launch_conv(stream, W_i_atom, WbA,   512, 133, 160);
  launch_conv(stream, W_i_bond, WbBnd, 512, 147, 160);
  launch_conv(stream, W_h,      WbH,   5 * 512, 512, 512);
  launch_conv(stream, W_lr,     WbLR,  512, 1536, 1536);
  launch_conv(stream, Wih_f,    WbIHf, 1536, 512, 512);
  launch_conv(stream, Whh_f,    WbHHf, 1536, 512, 512);
  launch_conv(stream, Wih_b,    WbIHb, 1536, 512, 512);
  launch_conv(stream, Whh_b,    WbHHb, 1536, 512, 512);
  launch_conv(stream, W_o,      WbO,   512, 1024, 1024);

  // ---- input projections (relu + f32 message copy) ----
  launch_gemm_t<true, false, false, true, false>(stream, fa_bf, 160, WbA,
      iatom, matom, nullptr, nullptr, nullptr, NA, 512, 160, 512);
  launch_gemm_t<true, false, false, true, false>(stream, fb_bf, 160, WbBnd,
      ibond, mbond, nullptr, nullptr, nullptr, NB, 512, 160, 512);

  // ---- message passing loop (relu + residual input_bond) ----
  for (int d = 0; d < 5; ++d) {
    agg_accum_kernel<<<gNA, TB, 0, stream>>>(mbond, a2b, matom, NAH);
    bond_pre_kernel<<<gNB, TB, 0, stream>>>(matom, mbond, b2a, b2revb, pre_bf, NBH);
    launch_gemm_t<true, true, false, false, false>(stream, pre_bf, 512,
        WbH + (size_t)d * 512 * 512, mbond, nullptr, ibond, nullptr, nullptr,
        NB, 512, 512, 512);
  }

  // ---- readout ----
  final_cat_kernel<<<gNA, TB, 0, stream>>>(mbond, a2b, matom, iatom, cat_bf, NAH);
  launch_gemm_t<false, false, false, false, false>(stream, cat_bf, 1536, WbLR,
      node, nullptr, nullptr, nullptr, nullptr, NA, 512, 1536, 512);
  postnode_kernel<<<gNA, TB, 0, stream>>>(node, gru_bias, msg_bf, gruc_bf, NAH);
  h0_kernel<<<gM, TB, 0, stream>>>(node, h_f, h_b, hbf_f, hbf_b, MH);

  // ---- bidirectional GRU (per-step gate GEMMs with bias + fused cell) ----
  for (int t = 0; t < 48; ++t) {  // forward
    launch_gemm_t<false, false, true, false, false>(stream, msg_bf + (size_t)t * 512,
        (long)48 * 512, WbIHf, gi, nullptr, nullptr, bih_f, nullptr,
        n_mols, 1536, 512, 1536);
    launch_gemm_t<false, false, true, false, false>(stream, hbf_f, 512, WbHHf,
        gh, nullptr, nullptr, bhh_f, nullptr, n_mols, 1536, 512, 1536);
    gru_combine_kernel<<<gM, TB, 0, stream>>>(gi, gh, h_f, hbf_f, gruc_bf, t, 0, MH);
  }
  for (int s = 0; s < 48; ++s) {  // backward
    int t = 47 - s;
    launch_gemm_t<false, false, true, false, false>(stream, msg_bf + (size_t)t * 512,
        (long)48 * 512, WbIHb, gi, nullptr, nullptr, bih_b, nullptr,
        n_mols, 1536, 512, 1536);
    launch_gemm_t<false, false, true, false, false>(stream, hbf_b, 512, WbHHb,
        gh, nullptr, nullptr, bhh_b, nullptr, n_mols, 1536, 512, 1536);
    gru_combine_kernel<<<gM, TB, 0, stream>>>(gi, gh, h_b, hbf_b, gruc_bf, t, 512, MH);
  }

  // ---- output projection (bias + relu) ----
  launch_gemm_t<true, false, true, false, false>(stream, gruc_bf, 1024, WbO,
      (float*)d_out, nullptr, nullptr, b_o, nullptr, NA, 512, 1024, 512);
}
